// QuantizedLinearWithLoRA_9380208575265
// MI455X (gfx1250) — matrix operand order repaired and checked by reading, not compiled, not run on hardware
//
#include <hip/hip_runtime.h>

#define IN_F   4096
#define OUT_F  4096
#define RANK   16
#define M_TOK  (4 * 2048)      // 8192 rows of x
#define LORA_SCALE 1.0f        // ALPHA / RANK = 16/16

typedef __attribute__((ext_vector_type(16))) __bf16 v16bf;
typedef __attribute__((ext_vector_type(8)))  float  v8f;

__device__ __forceinline__ unsigned short f2bf(float f) {
    unsigned int u = __float_as_uint(f);
    u += 0x7FFFu + ((u >> 16) & 1u);            // round-to-nearest-even
    return (unsigned short)(u >> 16);
}

// Low 32 bits of a flat pointer into LDS == LDS byte address (ISA §10.2 aperture map)
__device__ __forceinline__ unsigned lds_u32(const void* p) {
    return (unsigned)(unsigned long long)(uintptr_t)p;
}

// one 16-byte global -> LDS async copy (GV mode, per-lane addresses, ASYNCcnt)
#define ASYNC_CP_B128(ldsoff, gaddr)                                         \
    asm volatile("global_load_async_to_lds_b128 %0, %1, off"                 \
                 :: "v"(ldsoff), "v"(gaddr) : "memory")

// ---------------------------------------------------------------------------
// Kernel 1: W'[n][k] = (q-8)*scale + (ALPHA/RANK)*sum_r B[n][r]*A[r][k]  (bf16)
// ---------------------------------------------------------------------------
__global__ void __launch_bounds__(256) dequant_fold_lora(
    const int* __restrict__ q, const float* __restrict__ scales,
    const float* __restrict__ A, const float* __restrict__ B,
    unsigned short* __restrict__ Wt)
{
    long long base = ((long long)blockIdx.x * 256 + threadIdx.x) * 8;
    int n = (int)(base >> 12);                  // /4096 (same for all 8 elems)
    int k = (int)(base & 4095);
    float s = scales[base >> 6];                // block of 64: same for all 8 elems

    float Bn[RANK];
#pragma unroll
    for (int r = 0; r < RANK; ++r) Bn[r] = B[n * RANK + r];

    unsigned int packed[4];
#pragma unroll
    for (int j = 0; j < 8; j += 2) {
        float v0 = (float)(q[base + j]     - 8) * s;
        float v1 = (float)(q[base + j + 1] - 8) * s;
        float l0 = 0.f, l1 = 0.f;
#pragma unroll
        for (int r = 0; r < RANK; ++r) {
            l0 = fmaf(Bn[r], A[r * IN_F + k + j],     l0);
            l1 = fmaf(Bn[r], A[r * IN_F + k + j + 1], l1);
        }
        v0 = fmaf(LORA_SCALE, l0, v0);
        v1 = fmaf(LORA_SCALE, l1, v1);
        packed[j >> 1] = (unsigned int)f2bf(v0) | ((unsigned int)f2bf(v1) << 16);
    }
    *reinterpret_cast<uint4*>(Wt + base) =
        make_uint4(packed[0], packed[1], packed[2], packed[3]);
}

// ---------------------------------------------------------------------------
// Kernel 2: x f32 -> bf16 (so GEMM operands are raw-copyable via async-to-LDS,
// and x_bf16 (64MB) + W' (32MB) become L2-resident: 96MB < 192MB L2)
// ---------------------------------------------------------------------------
__global__ void __launch_bounds__(256) convert_x_bf16(
    const float* __restrict__ x, unsigned short* __restrict__ xb)
{
    long long base = ((long long)blockIdx.x * 256 + threadIdx.x) * 8;
    float4 a = *reinterpret_cast<const float4*>(x + base);
    float4 b = *reinterpret_cast<const float4*>(x + base + 4);
    uint4 o;
    o.x = (unsigned)f2bf(a.x) | ((unsigned)f2bf(a.y) << 16);
    o.y = (unsigned)f2bf(a.z) | ((unsigned)f2bf(a.w) << 16);
    o.z = (unsigned)f2bf(b.x) | ((unsigned)f2bf(b.y) << 16);
    o.w = (unsigned)f2bf(b.z) | ((unsigned)f2bf(b.w) << 16);
    *reinterpret_cast<uint4*>(xb + base) = o;
}

// ---------------------------------------------------------------------------
// GEMM tiling shared by both variants
// ---------------------------------------------------------------------------
#define BM 128
#define BN 128
#define BK 64
#define XS_STRIDE (BK + 8)   // 72 bf16 = 144 B: 16B-aligned rows + bank-conflict pad
#define WS_STRIDE (BK + 8)

// ---------------------------------------------------------------------------
// Kernel 3a: async double-buffered WMMA GEMM, last K-tile peeled so the
// steady-state loop is branch-free:
//   issue(next tile: 8x GLOBAL_LOAD_ASYNC_TO_LDS_B128) ->
//   s_wait_asynccnt 8 (per-wave ASYNCcnt, in-order ==> current tile resident) ->
//   barrier -> 16x v_wmma_f32_16x16x32_bf16 -> barrier
// ---------------------------------------------------------------------------
__global__ void __launch_bounds__(256) qlora_gemm_async(
    const unsigned short* __restrict__ xb, const unsigned short* __restrict__ Wt,
    const float* __restrict__ bias, float* __restrict__ out)
{
    __shared__ __align__(16) unsigned short xs [2][BM * XS_STRIDE];
    __shared__ __align__(16) unsigned short wsh[2][BN * WS_STRIDE];

    const int tid   = threadIdx.x;
    const int lane  = tid & 31;           // wave32
    const int wid   = tid >> 5;           // 0..7
    const int wm    = wid & 3;            // 4 waves along M, 32 rows each
    const int wn    = wid >> 2;           // 2 waves along N, 64 cols each
    const int m0    = blockIdx.y * BM;
    const int n0    = blockIdx.x * BN;
    const int lhalf = (lane < 16) ? 0 : 1;
    const int lmod  = lane & 15;

    // copy-slot mapping: id in [0,1024): row = id>>3 (128 rows), c = id&7 (8 x 16B)
    const int crow = tid >> 3;            // rows handled: crow, crow+32, crow+64, crow+96
    const int cc   = tid & 7;

    v8f acc[2][4] = {};

    auto issue_tile = [&](int kt, int buf) {
#pragma unroll
        for (int i = 0; i < 4; ++i) {
            int row = crow + i * 32;
            unsigned lx = lds_u32(&xs[buf][row * XS_STRIDE + cc * 8]);
            unsigned long long gx = (unsigned long long)(uintptr_t)
                (xb + (long long)(m0 + row) * IN_F + kt + cc * 8);
            ASYNC_CP_B128(lx, gx);
            unsigned lw = lds_u32(&wsh[buf][row * WS_STRIDE + cc * 8]);
            unsigned long long gw = (unsigned long long)(uintptr_t)
                (Wt + (long long)(n0 + row) * IN_F + kt + cc * 8);
            ASYNC_CP_B128(lw, gw);
        }
    };

    auto compute_tile = [&](int buf) {
#pragma unroll
        for (int kk = 0; kk < BK; kk += 32) {
            // A fragments (16x32 bf16): lanes 0-15 -> K[0..7],K[16..23]; lanes 16-31 -> +8
            union FragA { v16bf v; uint4 q[2]; } a[2];
#pragma unroll
            for (int ti = 0; ti < 2; ++ti) {
                int row = wm * 32 + ti * 16 + lmod;
                const unsigned short* pr = &xs[buf][row * XS_STRIDE + kk + lhalf * 8];
                a[ti].q[0] = *reinterpret_cast<const uint4*>(pr);       // K+0..7
                a[ti].q[1] = *reinterpret_cast<const uint4*>(pr + 16);  // K+16..23
            }
            // B fragments (32x16 bf16): lanes 0-15 -> K[0..7],K[16..23] at col n; lanes 16-31 -> +8
            union FragB { v16bf v; uint4 q[2]; } bf[4];
#pragma unroll
            for (int tj = 0; tj < 4; ++tj) {
                int row = wn * 64 + tj * 16 + lmod;                     // n index
                const unsigned short* pr = &wsh[buf][row * WS_STRIDE + kk + lhalf * 8];
                bf[tj].q[0] = *reinterpret_cast<const uint4*>(pr);      // K+0..7
                bf[tj].q[1] = *reinterpret_cast<const uint4*>(pr + 16); // K+16..23
            }
#pragma unroll
            for (int ti = 0; ti < 2; ++ti)
#pragma unroll
                for (int tj = 0; tj < 4; ++tj)
                    acc[ti][tj] = __builtin_amdgcn_wmma_f32_16x16x32_bf16(
                        false, a[ti].v, false, bf[tj].v,
                        (short)0, acc[ti][tj], false, false);
        }
    };

    issue_tile(0, 0);

    // steady state: 63 branch-free iterations with copy/compute overlap
    int kt = 0;
#pragma unroll 1
    for (; kt < IN_F - BK; kt += BK) {
        const int buf = (kt >> 6) & 1;    // kt / BK
        issue_tile(kt + BK, buf ^ 1);                     // overlap next copy
        asm volatile("s_wait_asynccnt 0x8" ::: "memory"); // cur tile's 8 done
        __syncthreads();
        compute_tile(buf);
        __syncthreads();                  // done reading buf before refill
    }
    // peeled final tile: nothing left to issue, drain fully, no trailing barrier
    {
        const int buf = (kt >> 6) & 1;
        asm volatile("s_wait_asynccnt 0x0" ::: "memory");
        __syncthreads();
        compute_tile(buf);
    }

    // epilogue: C/D layout (VGPR r -> rows r / r+8 by lane half, N = lane&15)
#pragma unroll
    for (int tj = 0; tj < 4; ++tj) {
        int col = n0 + wn * 64 + tj * 16 + lmod;
        float bv = bias[col];
#pragma unroll
        for (int ti = 0; ti < 2; ++ti) {
            int rbase = m0 + wm * 32 + ti * 16 + lhalf * 8;
#pragma unroll
            for (int r = 0; r < 8; ++r)
                out[(long long)(rbase + r) * OUT_F + col] = acc[ti][tj][r] + bv;
        }
    }
}

// ---------------------------------------------------------------------------
// Kernel 3b: fallback GEMM (on-the-fly f32->bf16 of x), used if ws can't hold
// both W' and x_bf16. Same math, single-buffered LDS.
// ---------------------------------------------------------------------------
__global__ void __launch_bounds__(256) qlora_gemm_sync(
    const float* __restrict__ x, const unsigned short* __restrict__ Wt,
    const float* __restrict__ bias, float* __restrict__ out)
{
    __shared__ __align__(16) unsigned short xs [BM * XS_STRIDE];
    __shared__ __align__(16) unsigned short wsh[BN * WS_STRIDE];

    const int tid   = threadIdx.x;
    const int lane  = tid & 31;
    const int wid   = tid >> 5;
    const int wm    = wid & 3;
    const int wn    = wid >> 2;
    const int m0    = blockIdx.y * BM;
    const int n0    = blockIdx.x * BN;
    const int lhalf = (lane < 16) ? 0 : 1;
    const int lmod  = lane & 15;

    v8f acc[2][4] = {};

    for (int kt = 0; kt < IN_F; kt += BK) {
#pragma unroll
        for (int i = 0; i < 8; ++i) {
            int id  = tid + i * 256;
            int row = id >> 4;
            int c4  = id & 15;
            float4 v = *reinterpret_cast<const float4*>(
                x + (long long)(m0 + row) * IN_F + kt + c4 * 4);
            unsigned int lo = (unsigned)f2bf(v.x) | ((unsigned)f2bf(v.y) << 16);
            unsigned int hi = (unsigned)f2bf(v.z) | ((unsigned)f2bf(v.w) << 16);
            unsigned int* p = reinterpret_cast<unsigned int*>(xs + row * XS_STRIDE + c4 * 4);
            p[0] = lo; p[1] = hi;
        }
#pragma unroll
        for (int i = 0; i < 4; ++i) {
            int id  = tid + i * 256;
            int row = id >> 3;
            int c   = id & 7;
            uint4 v = *reinterpret_cast<const uint4*>(
                Wt + (long long)(n0 + row) * IN_F + kt + c * 8);
            *reinterpret_cast<uint4*>(wsh + row * WS_STRIDE + c * 8) = v;
        }
        __syncthreads();

#pragma unroll
        for (int kk = 0; kk < BK; kk += 32) {
            union FragA { v16bf v; uint4 q[2]; } a[2];
#pragma unroll
            for (int ti = 0; ti < 2; ++ti) {
                int row = wm * 32 + ti * 16 + lmod;
                const unsigned short* pr = xs + row * XS_STRIDE + kk + lhalf * 8;
                a[ti].q[0] = *reinterpret_cast<const uint4*>(pr);
                a[ti].q[1] = *reinterpret_cast<const uint4*>(pr + 16);
            }
            union FragB { v16bf v; uint4 q[2]; } bf[4];
#pragma unroll
            for (int tj = 0; tj < 4; ++tj) {
                int row = wn * 64 + tj * 16 + lmod;
                const unsigned short* pr = wsh + row * WS_STRIDE + kk + lhalf * 8;
                bf[tj].q[0] = *reinterpret_cast<const uint4*>(pr);
                bf[tj].q[1] = *reinterpret_cast<const uint4*>(pr + 16);
            }
#pragma unroll
            for (int ti = 0; ti < 2; ++ti)
#pragma unroll
                for (int tj = 0; tj < 4; ++tj)
                    acc[ti][tj] = __builtin_amdgcn_wmma_f32_16x16x32_bf16(
                        false, a[ti].v, false, bf[tj].v,
                        (short)0, acc[ti][tj], false, false);
        }
        __syncthreads();
    }

#pragma unroll
    for (int tj = 0; tj < 4; ++tj) {
        int col = n0 + wn * 64 + tj * 16 + lmod;
        float bv = bias[col];
#pragma unroll
        for (int ti = 0; ti < 2; ++ti) {
            int rbase = m0 + wm * 32 + ti * 16 + lhalf * 8;
#pragma unroll
            for (int r = 0; r < 8; ++r)
                out[(long long)(rbase + r) * OUT_F + col] = acc[ti][tj][r] + bv;
        }
    }
}

extern "C" void kernel_launch(void* const* d_in, const int* in_sizes, int n_in,
                              void* d_out, int out_size, void* d_ws, size_t ws_size,
                              hipStream_t stream) {
    const float* x      = (const float*)d_in[0];
    const int*   q      = (const int*)  d_in[1];
    const float* scales = (const float*)d_in[2];
    const float* lora_A = (const float*)d_in[3];
    const float* lora_B = (const float*)d_in[4];
    const float* b      = (const float*)d_in[5];

    const size_t wt_bytes = (size_t)OUT_F * IN_F * sizeof(unsigned short); // 32 MB
    const size_t xb_bytes = (size_t)M_TOK * IN_F * sizeof(unsigned short); // 64 MB

    unsigned short* Wt = (unsigned short*)d_ws;

    // Pass 1: dequant + fold LoRA into bf16 W'
    dim3 g1(((long long)OUT_F * IN_F) / (256 * 8));   // 8192 blocks
    dequant_fold_lora<<<g1, dim3(256), 0, stream>>>(q, scales, lora_A, lora_B, Wt);

    dim3 g2(OUT_F / BN, M_TOK / BM);                  // 32 x 64

    if (ws_size >= wt_bytes + xb_bytes) {
        // Pass 2: x -> bf16, then async double-buffered WMMA GEMM
        unsigned short* xbf = (unsigned short*)((char*)d_ws + wt_bytes);
        dim3 gc(((long long)M_TOK * IN_F) / (256 * 8));  // 16384 blocks
        convert_x_bf16<<<gc, dim3(256), 0, stream>>>(x, xbf);
        qlora_gemm_async<<<g2, dim3(256), 0, stream>>>(xbf, Wt, b, (float*)d_out);
    } else {
        qlora_gemm_sync<<<g2, dim3(256), 0, stream>>>(x, Wt, b, (float*)d_out);
    }
}
